// PoincareResidualBlock_34522947125770
// MI455X (gfx1250) — matrix operand order repaired and checked by reading, not compiled, not run on hardware
//
#include <hip/hip_runtime.h>
#include <cmath>

#define EPSF 1e-15f
#define MAXN 0.99999f
#define NB 16
#define NC 256
#define HP 34
#define NPIX 16384
#define KTOT 2304

typedef __attribute__((ext_vector_type(8)))  float  v8f;
typedef __attribute__((ext_vector_type(16))) __bf16 v16bf;
typedef __attribute__((ext_vector_type(8)))  __bf16 v8bf;

static __device__ __forceinline__ unsigned short f2bf(float f){
  unsigned int u = __float_as_uint(f);
  unsigned int r = (u + 0x7FFFu + ((u >> 16) & 1u)) >> 16;
  return (unsigned short)r;
}
static __device__ __forceinline__ float my_tanh(float x){
  float ax = fminf(fabsf(x), 15.0f);
  float e = expf(2.0f*ax);
  float r = (e - 1.0f)/(e + 1.0f);
  return x < 0.0f ? -r : r;
}
static __device__ __forceinline__ float my_atanh(float x){
  x = fminf(fmaxf(x, -1.0f + 1e-7f), 1.0f - 1e-7f);
  return 0.5f*logf((1.0f + x)/(1.0f - x));
}
static __device__ __forceinline__ float my_asinh(float x){
  float ax = fabsf(x);
  float r = logf(ax + sqrtf(ax*ax + 1.0f));
  return x < 0.0f ? -r : r;
}
static __device__ __forceinline__ float my_sinh(float x){
  float ax = fabsf(x);
  float e = expf(ax);
  float r = 0.5f*(e - 1.0f/e);
  return x < 0.0f ? -r : r;
}
static __device__ __forceinline__ float wred32(float v){
  v += __shfl_xor(v, 16, 32);
  v += __shfl_xor(v, 8, 32);
  v += __shfl_xor(v, 4, 32);
  v += __shfl_xor(v, 2, 32);
  v += __shfl_xor(v, 1, 32);
  return v;
}
static __device__ __forceinline__ float block_reduce_256(float v, float* red){
  int t = threadIdx.x;
  red[t] = v; __syncthreads();
  for (int s = 128; s > 0; s >>= 1){
    if (t < s) red[t] += red[t + s];
    __syncthreads();
  }
  float r = red[0];
  __syncthreads();
  return r;
}

// ---------------- utility kernels ----------------
__global__ __launch_bounds__(256) void k_zero(unsigned int* p, unsigned int n){
  unsigned int i = blockIdx.x*256u + threadIdx.x;
  unsigned int st = gridDim.x*256u;
  for (; i < n; i += st) p[i] = 0u;
}

// z [K,N] f32 -> zt [N,K] bf16
__global__ __launch_bounds__(256) void k_prep_zt(const float* __restrict__ z,
                                                 unsigned short* __restrict__ zt){
  int i = blockIdx.x*256 + threadIdx.x;
  if (i >= KTOT*NC) return;
  int n = i / KTOT;
  int k = i - n*KTOT;
  zt[i] = f2bf(z[k*NC + n]);
}

__global__ __launch_bounds__(256) void k_znorm(const float* __restrict__ z,
                                               float* __restrict__ zn){
  int j = threadIdx.x;
  float s = 0.f;
  for (int k = 0; k < KTOT; ++k){ float v = z[(size_t)k*NC + j]; s += v*v; }
  zn[j] = sqrtf(fmaxf(s, EPSF));
}

// x [B,C,H,W] -> vpad bf16 [B,34,34,C] (logmap0*scale), spad (row sqnorm), xl [p,C] copy
__global__ __launch_bounds__(256) void k_stage_x(const float* __restrict__ x,
    unsigned short* __restrict__ vpad, float* __restrict__ spad,
    float* __restrict__ xl, float scale){
  int p = blockIdx.x*256 + threadIdx.x;
  int b = p >> 10, hw = p & 1023;
  int h = hw >> 5, w = hw & 31;
  const float* base = x + (size_t)b*NC*1024 + hw;
  float s = 0.f;
  for (int ch = 0; ch < NC; ++ch){ float v = base[(size_t)ch*1024]; s += v*v; }
  float n = sqrtf(fmaxf(s, EPSF));
  float st = scale * my_atanh(n) / n;
  size_t vo = ((size_t)(b*HP + h + 1)*HP + (w + 1))*NC;
  float* xr = xl + (size_t)p*NC;
  for (int ch = 0; ch < NC; ++ch){
    float v = base[(size_t)ch*1024];
    xr[ch] = v;
    vpad[vo + ch] = f2bf(st*v);
  }
  spad[(size_t)(b*HP + h + 1)*HP + (w + 1)] = st*st*s;
}

// channel-last input -> vpad/spad (one wave per pixel)
__global__ __launch_bounds__(256) void k_stage_h(const float* __restrict__ in,
    unsigned short* __restrict__ vpad, float* __restrict__ spad, float scale){
  const int lane = threadIdx.x & 31, wv = threadIdx.x >> 5;
  const int p = blockIdx.x*8 + wv;
  const int b = p >> 10, hw = p & 1023, h = hw >> 5, w = hw & 31;
  const float* row = in + (size_t)p*NC;
  float xv[8]; float s = 0.f;
  #pragma unroll
  for (int j = 0; j < 8; ++j){ xv[j] = row[lane + 32*j]; s += xv[j]*xv[j]; }
  s = wred32(s);
  float n = sqrtf(fmaxf(s, EPSF));
  float st = scale * my_atanh(n) / n;
  size_t vo = ((size_t)(b*HP + h + 1)*HP + (w + 1))*NC;
  #pragma unroll
  for (int j = 0; j < 8; ++j) vpad[vo + lane + 32*j] = f2bf(st*xv[j]);
  if (lane == 0) spad[(size_t)(b*HP + h + 1)*HP + (w + 1)] = st*st*s;
}

// 3x3 box-sum of per-pixel tangent sqnorms -> patch sqnorm
__global__ __launch_bounds__(256) void k_patchnorm(const float* __restrict__ spad,
                                                   float* __restrict__ pn){
  int p = blockIdx.x*256 + threadIdx.x;
  int b = p >> 10, hw = p & 1023, h = hw >> 5, w = hw & 31;
  float s = 0.f;
  #pragma unroll
  for (int di = 0; di < 3; ++di)
    #pragma unroll
    for (int dj = 0; dj < 3; ++dj)
      s += spad[(size_t)(b*HP + h + di)*HP + (w + dj)];
  pn[p] = s;
}

// ---------------- implicit-GEMM conv3x3 + Poincare-FC epilogue ----------------
// block = 8 waves; block tile = 64 pixels x 256 outputs
// wave  = 32 pixels (2 WMMA M-subtiles) x 64 outputs (4 N-tiles) -> 8 accumulators
// Each B fragment feeds 2 WMMAs (reused across M) -> 1.5 b128 loads per WMMA.
__global__ __launch_bounds__(256) void k_gemm_fc(
    const unsigned short* __restrict__ vpad,
    const unsigned short* __restrict__ zt,     // [N=256, K=2304] bf16
    const float* __restrict__ znorm,           // [256]
    const float* __restrict__ pn,              // [16384] patch sqnorm
    float* __restrict__ out)                   // [16384, 256]
{
  __shared__ float rowsum[64];
  const int tid = threadIdx.x;
  const int lane = tid & 31;
  const int wv = tid >> 5;
  const int mGroup = wv & 1;        // which 32-pixel half of the block tile
  const int nGroup = wv >> 1;       // which 64-column group
  const int half = lane >> 4;
  const int l16 = lane & 15;

  const int pixbase = blockIdx.x*64;
  // two A-row pixels per lane position (M-subtiles 0 and 1 of this wave)
  const int p0 = pixbase + mGroup*32 + l16;
  const int p1 = p0 + 16;
  const int b0 = p0 >> 10, h0 = (p0 >> 5) & 31, w0 = p0 & 31;
  const int b1 = p1 >> 10, h1 = (p1 >> 5) & 31, w1 = p1 & 31;

  v8f acc[2][4];
  #pragma unroll
  for (int m = 0; m < 2; ++m)
    #pragma unroll
    for (int t = 0; t < 4; ++t)
      #pragma unroll
      for (int r = 0; r < 8; ++r) acc[m][t][r] = 0.0f;

  const unsigned short* zc[4];
  #pragma unroll
  for (int t = 0; t < 4; ++t){
    int col = nGroup*64 + t*16 + l16;
    zc[t] = zt + (size_t)col*KTOT + 8*half;
  }

  for (int tap = 0; tap < 9; ++tap){
    const int di = tap/3, dj = tap - di*3;
    const unsigned short* arow0 =
        vpad + ((size_t)((b0*HP + h0 + di)*HP + (w0 + dj)))*NC + 8*half;
    const unsigned short* arow1 =
        vpad + ((size_t)((b1*HP + h1 + di)*HP + (w1 + dj)))*NC + 8*half;
    if (tap < 8){
      const int di2 = (tap+1)/3, dj2 = (tap+1) - di2*3;
      __builtin_prefetch(vpad + ((size_t)((b0*HP + h0 + di2)*HP + (w0 + dj2)))*NC, 0, 1);
      __builtin_prefetch(vpad + ((size_t)((b1*HP + h1 + di2)*HP + (w1 + dj2)))*NC, 0, 1);
    }
    const int kbase = tap*NC;
    #pragma unroll
    for (int kk = 0; kk < 8; ++kk){
      const int ch0 = kk*32;
      // A fragments (16-bit A 16x32 layout: lanes0-15 K0-7/16-23, lanes16-31 K8-15/24-31)
      v8bf a0lo = *reinterpret_cast<const v8bf*>(arow0 + ch0);
      v8bf a0hi = *reinterpret_cast<const v8bf*>(arow0 + ch0 + 16);
      v8bf a1lo = *reinterpret_cast<const v8bf*>(arow1 + ch0);
      v8bf a1hi = *reinterpret_cast<const v8bf*>(arow1 + ch0 + 16);
      v16bf a0, a1;
      #pragma unroll
      for (int i = 0; i < 8; ++i){
        a0[i] = a0lo[i]; a0[8+i] = a0hi[i];
        a1[i] = a1lo[i]; a1[8+i] = a1hi[i];
      }
      const int k0 = kbase + ch0;
      #pragma unroll
      for (int t = 0; t < 4; ++t){
        // B fragment (lane n = col; lanes0-15 K0-7/16-23, lanes16-31 K8-15/24-31)
        v8bf blo = *reinterpret_cast<const v8bf*>(zc[t] + k0);
        v8bf bhi = *reinterpret_cast<const v8bf*>(zc[t] + k0 + 16);
        v16bf bb;
        #pragma unroll
        for (int i = 0; i < 8; ++i){ bb[i] = blo[i]; bb[8+i] = bhi[i]; }
        acc[0][t] = __builtin_amdgcn_wmma_f32_16x16x32_bf16(
            false, a0, false, bb, (short)0, acc[0][t], false, false);
        acc[1][t] = __builtin_amdgcn_wmma_f32_16x16x32_bf16(
            false, a1, false, bb, (short)0, acc[1][t], false, false);
      }
    }
  }

  if (tid < 64) rowsum[tid] = 0.0f;
  __syncthreads();

  // per-row factor f = lam(u) * tanh(n)/n  (fold of expmap0 + lambda)
  float fA[2][8];
  #pragma unroll
  for (int m = 0; m < 2; ++m)
    #pragma unroll
    for (int r = 0; r < 8; ++r){
      float s = pn[pixbase + mGroup*32 + m*16 + half*8 + r];
      float n = sqrtf(fmaxf(s, EPSF));
      float tn = fminf(my_tanh(n), MAXN);
      fA[m][r] = 2.0f*tn/(fmaxf(1.0f - tn*tn, EPSF)*n);
    }
  float zn[4];
  #pragma unroll
  for (int t = 0; t < 4; ++t) zn[t] = znorm[nGroup*64 + t*16 + l16];

  float part[2][8];
  #pragma unroll
  for (int m = 0; m < 2; ++m)
    #pragma unroll
    for (int r = 0; r < 8; ++r) part[m][r] = 0.0f;

  #pragma unroll
  for (int m = 0; m < 2; ++m)
    #pragma unroll
    for (int t = 0; t < 4; ++t)
      #pragma unroll
      for (int r = 0; r < 8; ++r){
        float arg = fA[m][r]*acc[m][t][r]/zn[t];
        float d = 2.0f*zn[t]*my_asinh(arg);
        float y = my_sinh(d);
        acc[m][t][r] = y;
        part[m][r] += y*y;
      }

  // sum over 16 lanes of same half (columns of this wave)
  #pragma unroll
  for (int m = 0; m < 2; ++m)
    #pragma unroll
    for (int r = 0; r < 8; ++r){
      float v = part[m][r];
      v += __shfl_xor(v, 1, 32);
      v += __shfl_xor(v, 2, 32);
      v += __shfl_xor(v, 4, 32);
      v += __shfl_xor(v, 8, 32);
      part[m][r] = v;
    }
  if (l16 == 0){
    #pragma unroll
    for (int m = 0; m < 2; ++m)
      #pragma unroll
      for (int r = 0; r < 8; ++r)
        atomicAdd(&rowsum[mGroup*32 + m*16 + half*8 + r], part[m][r]);
  }
  __syncthreads();

  #pragma unroll
  for (int m = 0; m < 2; ++m)
    #pragma unroll
    for (int t = 0; t < 4; ++t){
      const int col = nGroup*64 + t*16 + l16;
      #pragma unroll
      for (int r = 0; r < 8; ++r){
        const int row = mGroup*32 + m*16 + half*8 + r;
        float sc = 1.0f/(1.0f + sqrtf(1.0f + rowsum[row]));
        out[(size_t)(pixbase + row)*NC + col] = acc[m][t][r]*sc;
      }
    }
}

// ---------------- hyperbolic batch norm ----------------
__global__ __launch_bounds__(256) void k_bn_reduce1(const float* __restrict__ x,
    float* __restrict__ m_num, float* __restrict__ lam_den){
  __shared__ float chsum[256];
  __shared__ float lsum;
  const int tid = threadIdx.x, lane = tid & 31, wv = tid >> 5;
  chsum[tid] = 0.0f;
  if (tid == 0) lsum = 0.0f;
  __syncthreads();
  for (int it = 0; it < 32; ++it){
    int p = blockIdx.x*256 + it*8 + wv;
    const float* row = x + (size_t)p*NC;
    float xv[8]; float s = 0.f;
    #pragma unroll
    for (int j = 0; j < 8; ++j){ xv[j] = row[lane + 32*j]; s += xv[j]*xv[j]; }
    s = wred32(s);
    float lam = 2.0f/fmaxf(1.0f - s, EPSF);
    #pragma unroll
    for (int j = 0; j < 8; ++j) atomicAdd(&chsum[lane + 32*j], lam*xv[j]);
    if (lane == 0) atomicAdd(&lsum, lam - 1.0f);
  }
  __syncthreads();
  atomicAdd(&m_num[tid], chsum[tid]);
  if (tid == 0) atomicAdd(lam_den, lsum);
}

__global__ __launch_bounds__(256) void k_bn_stats(const float* __restrict__ m_num,
    const float* __restrict__ lam_den, const float* __restrict__ bias,
    float* __restrict__ mu, float* __restrict__ stats){
  __shared__ float red[256];
  int t = threadIdx.x;
  float m = m_num[t]/fmaxf(lam_den[0], EPSF);
  float nm2 = block_reduce_256(m*m, red);
  float nm = sqrtf(fmaxf(nm2, EPSF));
  float k = my_tanh(0.5f*my_atanh(nm))/nm;      // mobius_scalar_mul(0.5, m)
  float muv = k*m;
  mu[t] = muv;
  float bv = bias[t];
  float mu2 = block_reduce_256(muv*muv, red);
  float b2  = block_reduce_256(bv*bv, red);
  float mb  = block_reduce_256(muv*bv, red);
  if (t == 0){ stats[0] = mu2; stats[1] = b2; stats[2] = mb; }
}

__global__ __launch_bounds__(256) void k_bn_var(const float* __restrict__ x,
    const float* __restrict__ mu, const float* __restrict__ stats,
    float* __restrict__ var_acc){
  __shared__ float bsum[8];
  const int lane = threadIdx.x & 31, wv = threadIdx.x >> 5;
  const int p = blockIdx.x*8 + wv;
  const float* row = x + (size_t)p*NC;
  float s = 0.f, d = 0.f;
  #pragma unroll
  for (int j = 0; j < 8; ++j){
    float xv = row[lane + 32*j];
    float mv = mu[lane + 32*j];
    s += xv*xv; d += xv*mv;
  }
  float px2 = wred32(s);
  float xy  = wred32(d);
  float mu2 = stats[0];
  float A = 1.f - 2.f*xy + mu2;          // mobius_add(-x, mu)
  float B = 1.f - px2;
  float den = fmaxf(1.f - 2.f*xy + px2*mu2, EPSF);
  float num2 = A*A*px2 + B*B*mu2 - 2.f*A*B*xy;
  float nm = sqrtf(fmaxf(num2, EPSF))/den;
  float dist = 2.f*my_atanh(nm);
  if (lane == 0) bsum[wv] = dist*dist;
  __syncthreads();
  if (threadIdx.x == 0){
    float t = 0.f;
    for (int i = 0; i < 8; ++i) t += bsum[i];
    atomicAdd(var_acc, t);
  }
}

__global__ __launch_bounds__(256) void k_bn_apply(const float* __restrict__ x,
    const float* __restrict__ mu, const float* __restrict__ bias,
    const float* __restrict__ stats, const float* __restrict__ var_acc,
    const float* __restrict__ weight, float* __restrict__ out){
  const int lane = threadIdx.x & 31, wv = threadIdx.x >> 5;
  const int p = blockIdx.x*8 + wv;
  const float* row = x + (size_t)p*NC;
  float xv[8], muv[8], bv[8];
  float s0 = 0.f, s1 = 0.f, s2 = 0.f;
  #pragma unroll
  for (int j = 0; j < 8; ++j){
    int ch = lane + 32*j;
    xv[j] = row[ch]; muv[j] = mu[ch]; bv[j] = bias[ch];
    s0 += xv[j]*xv[j]; s1 += xv[j]*muv[j]; s2 += xv[j]*bv[j];
  }
  float px2 = wred32(s0), xy = wred32(s1), bx = wred32(s2);
  float mu2 = stats[0], b2 = stats[1], mb = stats[2];
  float var = var_acc[0]*(1.0f/16384.0f);
  float lam_mu = 2.f/fmaxf(1.f - mu2, EPSF);
  float lam_b  = 2.f/fmaxf(1.f - b2, EPSF);
  // v = logmap(mu, x):  madd = al*mu + be*x
  float Aa = 1.f - 2.f*xy + px2;
  float Bb = 1.f - mu2;
  float dm = fmaxf(1.f - 2.f*xy + mu2*px2, EPSF);
  float al = -Aa/dm, be = Bb/dm;
  float nm2 = al*al*mu2 + be*be*px2 + 2.f*al*be*xy;
  float nm = sqrtf(fmaxf(nm2, EPSF));
  float sv = (2.f/lam_mu)*my_atanh(nm)/nm;
  float vm = sv*al, vx = sv*be;
  // gyration(bias, -mu, v)
  float uw = vm*mb + vx*bx;
  float vw = -(vm*mu2 + vx*xy);
  float uv = -mb;
  float ag = -uw*mu2 - vw + 2.f*uv*vw;
  float bg = -vw*b2 + uw;
  float dg = fmaxf(1.f + 2.f*uv + b2*mu2, EPSF);
  float gx = vx;
  float gm = vm - 2.f*bg/dg;
  float gb = 2.f*ag/dg;
  float scl = (lam_mu/lam_b)*sqrtf(weight[0]/fmaxf(var, EPSF));
  float P = gx*scl, Q = gm*scl, R = gb*scl;   // u = P*x + Q*mu + R*bias
  float nu2 = P*P*px2 + Q*Q*mu2 + R*R*b2 + 2.f*(P*Q*xy + P*R*bx + Q*R*mb);
  float nu = sqrtf(fmaxf(nu2, EPSF));
  float ss = my_tanh(0.5f*lam_b*nu)/nu;       // expmap(bias, u): second term
  float Ps = P*ss, Qs = Q*ss, Rs = R*ss;
  float sec2 = ss*ss*nu2;
  float bsd = Ps*bx + Qs*mb + Rs*b2;
  float nA = 1.f + 2.f*bsd + sec2;
  float nB = 1.f - b2;
  float dd = fmaxf(1.f + 2.f*bsd + b2*sec2, EPSF);
  float cx = (nB/dd)*Ps;
  float cm = (nB/dd)*Qs;
  float cb = nA/dd + (nB/dd)*Rs;
  float on2 = cx*cx*px2 + cm*cm*mu2 + cb*cb*b2
            + 2.f*(cx*cm*xy + cx*cb*bx + cm*cb*mb);
  float on = sqrtf(fmaxf(on2, EPSF));
  float pr = on > MAXN ? MAXN/on : 1.0f;      // project
  cx *= pr; cm *= pr; cb *= pr;
  float* orow = out + (size_t)p*NC;
  #pragma unroll
  for (int j = 0; j < 8; ++j)
    orow[lane + 32*j] = cx*xv[j] + cm*muv[j] + cb*bv[j];
}

__global__ __launch_bounds__(256) void k_mobius_add(const float* __restrict__ xh,
    const float* __restrict__ yres, float* __restrict__ out){
  const int lane = threadIdx.x & 31, wv = threadIdx.x >> 5;
  const int p = blockIdx.x*8 + wv;
  const float* xr = xh + (size_t)p*NC;
  const float* yr = yres + (size_t)p*NC;
  float xv[8], yv[8];
  float sx = 0.f, sy = 0.f, sxy = 0.f;
  #pragma unroll
  for (int j = 0; j < 8; ++j){
    int ch = lane + 32*j;
    xv[j] = xr[ch]; yv[j] = yr[ch];
    sx += xv[j]*xv[j]; sy += yv[j]*yv[j]; sxy += xv[j]*yv[j];
  }
  float x2 = wred32(sx), y2 = wred32(sy), xy = wred32(sxy);
  float A = 1.f + 2.f*xy + y2;
  float B = 1.f - x2;
  float den = fmaxf(1.f + 2.f*xy + x2*y2, EPSF);
  float ia = A/den, ib = B/den;
  float* orow = out + (size_t)p*NC;
  #pragma unroll
  for (int j = 0; j < 8; ++j)
    orow[lane + 32*j] = ia*xv[j] + ib*yv[j];
}

__global__ __launch_bounds__(256) void k_hrelu(const float* __restrict__ in,
    float* __restrict__ out, int transposed){
  const int lane = threadIdx.x & 31, wv = threadIdx.x >> 5;
  const int p = blockIdx.x*8 + wv;
  const float* row = in + (size_t)p*NC;
  float xv[8]; float s = 0.f;
  #pragma unroll
  for (int j = 0; j < 8; ++j){ xv[j] = row[lane + 32*j]; s += xv[j]*xv[j]; }
  float px2 = wred32(s);
  float n = sqrtf(fmaxf(px2, EPSF));
  float t = my_atanh(n)/n;
  float sr = 0.f; float rv[8];
  #pragma unroll
  for (int j = 0; j < 8; ++j){ rv[j] = fmaxf(xv[j], 0.f); sr += rv[j]*rv[j]; }
  sr = wred32(sr);
  float nu = sqrtf(fmaxf(t*t*sr, EPSF));
  float k = fminf(my_tanh(nu), MAXN)/nu;
  float kt = k*t;
  if (!transposed){
    float* orow = out + (size_t)p*NC;
    #pragma unroll
    for (int j = 0; j < 8; ++j) orow[lane + 32*j] = kt*rv[j];
  } else {
    int b = p >> 10, hw = p & 1023;
    #pragma unroll
    for (int j = 0; j < 8; ++j){
      int ch = lane + 32*j;
      out[((size_t)b*NC + ch)*1024 + hw] = kt*rv[j];
    }
  }
}

// ---------------- host launcher ----------------
extern "C" void kernel_launch(void* const* d_in, const int* in_sizes, int n_in,
                              void* d_out, int out_size, void* d_ws, size_t ws_size,
                              hipStream_t stream)
{
  (void)in_sizes; (void)n_in; (void)out_size; (void)ws_size;
  const float* x  = (const float*)d_in[0];
  const float* z1 = (const float*)d_in[1];
  const float* z2 = (const float*)d_in[2];
  const float* w1 = (const float*)d_in[3];
  const float* b1 = (const float*)d_in[4];
  const float* w2 = (const float*)d_in[5];
  const float* b2 = (const float*)d_in[6];
  float* outp = (float*)d_out;

  char* base = (char*)d_ws;
  size_t off = 0;
  auto alloc = [&](size_t bytes) -> void* {
    void* p = base + off;
    off += (bytes + 255) & ~(size_t)255;
    return p;
  };
  unsigned short* vpad = (unsigned short*)alloc((size_t)NB*HP*HP*NC*2);
  float* spad = (float*)alloc((size_t)NB*HP*HP*4);
  float* pn   = (float*)alloc((size_t)NPIX*4);
  unsigned short* zt1 = (unsigned short*)alloc((size_t)KTOT*NC*2);
  unsigned short* zt2 = (unsigned short*)alloc((size_t)KTOT*NC*2);
  float* zn1 = (float*)alloc(NC*4);
  float* zn2 = (float*)alloc(NC*4);
  float* xl  = (float*)alloc((size_t)NPIX*NC*4);
  float* hA  = (float*)alloc((size_t)NPIX*NC*4);
  float* hB  = (float*)alloc((size_t)NPIX*NC*4);
  float* m_num   = (float*)alloc(NC*4);   // these five are contiguous: 2816 B
  float* lam_den = (float*)alloc(256);
  float* mu      = (float*)alloc(NC*4);
  float* stats   = (float*)alloc(256);
  float* var_acc = (float*)alloc(256);

  // beta(9C/2,1/2)/beta(C/2,1/2)
  double bn_ = exp(lgamma(9.0*NC/2.0) + lgamma(0.5) - lgamma(9.0*NC/2.0 + 0.5));
  double bd_ = exp(lgamma((double)NC/2.0) + lgamma(0.5) - lgamma((double)NC/2.0 + 0.5));
  float scale = (float)(bn_/bd_);

  dim3 blk(256);
  k_zero<<<dim3(1024), blk, 0, stream>>>((unsigned int*)vpad,
      (unsigned)((size_t)NB*HP*HP*NC*2/4));
  k_zero<<<dim3(128), blk, 0, stream>>>((unsigned int*)spad, (unsigned)(NB*HP*HP));

  k_prep_zt<<<dim3(KTOT), blk, 0, stream>>>(z1, zt1);
  k_prep_zt<<<dim3(KTOT), blk, 0, stream>>>(z2, zt2);
  k_znorm<<<dim3(1), blk, 0, stream>>>(z1, zn1);
  k_znorm<<<dim3(1), blk, 0, stream>>>(z2, zn2);

  // conv1
  k_stage_x<<<dim3(64), blk, 0, stream>>>(x, vpad, spad, xl, scale);
  k_patchnorm<<<dim3(64), blk, 0, stream>>>(spad, pn);
  k_gemm_fc<<<dim3(256), blk, 0, stream>>>(vpad, zt1, zn1, pn, hA);
  // bn1
  k_zero<<<dim3(4), blk, 0, stream>>>((unsigned int*)m_num, 704u);
  k_bn_reduce1<<<dim3(64), blk, 0, stream>>>(hA, m_num, lam_den);
  k_bn_stats<<<dim3(1), blk, 0, stream>>>(m_num, lam_den, b1, mu, stats);
  k_bn_var<<<dim3(2048), blk, 0, stream>>>(hA, mu, stats, var_acc);
  k_bn_apply<<<dim3(2048), blk, 0, stream>>>(hA, mu, b1, stats, var_acc, w1, hB);
  k_hrelu<<<dim3(2048), blk, 0, stream>>>(hB, hA, 0);
  // conv2
  k_stage_h<<<dim3(2048), blk, 0, stream>>>(hA, vpad, spad, scale);
  k_patchnorm<<<dim3(64), blk, 0, stream>>>(spad, pn);
  k_gemm_fc<<<dim3(256), blk, 0, stream>>>(vpad, zt2, zn2, pn, hB);
  // bn2
  k_zero<<<dim3(4), blk, 0, stream>>>((unsigned int*)m_num, 704u);
  k_bn_reduce1<<<dim3(64), blk, 0, stream>>>(hB, m_num, lam_den);
  k_bn_stats<<<dim3(1), blk, 0, stream>>>(m_num, lam_den, b2, mu, stats);
  k_bn_var<<<dim3(2048), blk, 0, stream>>>(hB, mu, stats, var_acc);
  k_bn_apply<<<dim3(2048), blk, 0, stream>>>(hB, mu, b2, stats, var_acc, w2, hA);
  // residual + final relu (writes [B,C,H,W])
  k_mobius_add<<<dim3(2048), blk, 0, stream>>>(hA, xl, hB);
  k_hrelu<<<dim3(2048), blk, 0, stream>>>(hB, outp, 1);
}
